// CkyLinear_82626580840596
// MI455X (gfx1250) — matrix operand order repaired and checked by reading, not compiled, not run on hardware
//
#include <hip/hip_runtime.h>

// CDNA5 / gfx1250: wave32, WMMA 16x16x32 bf16 with f32 accumulate.
typedef __attribute__((ext_vector_type(16))) __bf16 v16bf;
typedef __attribute__((ext_vector_type(2)))  __bf16 v2bf;
typedef __attribute__((ext_vector_type(8)))  float  v8f;

#define OUT_F 4096
#define IN_F  4096
#define MTOT  16384      // B*S = 8*2048
#define MT    128        // M tile per workgroup
#define NT    128        // N tile per workgroup
#define KTILE 32         // K step (bf16 WMMA K)
#define LDP   40         // padded LDS row pitch in bf16 elems (16B aligned, 20-bank stride)

__device__ __forceinline__ unsigned pk(float lo, float hi) {
#if __has_builtin(__builtin_amdgcn_cvt_pk_bf16_f32)
  // Single-instruction packed f32 -> 2x bf16 (RNE): v_cvt_pk_bf16_f32
  union { v2bf v; unsigned u; } cv;
  cv.v = __builtin_amdgcn_cvt_pk_bf16_f32(lo, hi);
  return cv.u;
#else
  // Manual round-to-nearest-even fallback
  unsigned ul = __float_as_uint(lo);
  ul += 0x7FFFu + ((ul >> 16) & 1u);
  unsigned uh = __float_as_uint(hi);
  uh += 0x7FFFu + ((uh >> 16) & 1u);
  return (ul >> 16) | (uh & 0xFFFF0000u);
#endif
}

union Frag { unsigned u[8]; v16bf v; };

__global__ __launch_bounds__(256)
void qlinear_wmma_bf16(const float* __restrict__ x,
                       const int*   __restrict__ Wq,
                       const float* __restrict__ scale,
                       const float* __restrict__ zero,
                       const float* __restrict__ bias,
                       float*       __restrict__ out) {
  __shared__ __align__(16) unsigned short As[2][MT][LDP];   // x tile, bf16
  __shared__ __align__(16) unsigned short Bs[2][NT][LDP];   // dequantized W tile, bf16

  const int tid    = threadIdx.x;
  const int m_base = blockIdx.y * MT;
  const int n_base = blockIdx.x * NT;

  // Global staging: each thread owns half a row (16 elems) of both tiles.
  const int r  = tid >> 1;           // 0..127
  const int c0 = (tid & 1) * 16;     // 0 or 16

  const size_t x_row = (size_t)(m_base + r) * IN_F;
  const int    o     = n_base + r;
  // W_q flat[(o>>6)*64 + (o&63)][i]; scale/zero at (o&63)*4096 + i
  const size_t q_row = (size_t)((o >> 6) * 64 + (o & 63)) * IN_F;
  const size_t s_row = (size_t)(o & 63) * IN_F;

  float4 xa[4];   // staged x (fp32)
  float4 wd[4];   // staged dequantized W (fp32)

  auto load_tile = [&](int kt) {
    const int k0 = kt * KTILE + c0;
    const float4* xp = (const float4*)(x + x_row + k0);
    #pragma unroll
    for (int j = 0; j < 4; ++j) xa[j] = xp[j];
    const int4*   qp = (const int4*)  (Wq    + q_row + k0);
    const float4* sp = (const float4*)(scale + s_row + k0);
    const float4* zp = (const float4*)(zero  + s_row + k0);
    #pragma unroll
    for (int j = 0; j < 4; ++j) {
      const int4 q = qp[j]; const float4 s = sp[j]; const float4 z = zp[j];
      wd[j].x = ((float)q.x - z.x) * s.x;
      wd[j].y = ((float)q.y - z.y) * s.y;
      wd[j].z = ((float)q.z - z.z) * s.z;
      wd[j].w = ((float)q.w - z.w) * s.w;
    }
  };

  auto store_tile = [&](int buf) {
    uint4 a0 = { pk(xa[0].x, xa[0].y), pk(xa[0].z, xa[0].w),
                 pk(xa[1].x, xa[1].y), pk(xa[1].z, xa[1].w) };
    uint4 a1 = { pk(xa[2].x, xa[2].y), pk(xa[2].z, xa[2].w),
                 pk(xa[3].x, xa[3].y), pk(xa[3].z, xa[3].w) };
    *(uint4*)&As[buf][r][c0]     = a0;
    *(uint4*)&As[buf][r][c0 + 8] = a1;
    uint4 b0 = { pk(wd[0].x, wd[0].y), pk(wd[0].z, wd[0].w),
                 pk(wd[1].x, wd[1].y), pk(wd[1].z, wd[1].w) };
    uint4 b1 = { pk(wd[2].x, wd[2].y), pk(wd[2].z, wd[2].w),
                 pk(wd[3].x, wd[3].y), pk(wd[3].z, wd[3].w) };
    *(uint4*)&Bs[buf][r][c0]     = b0;
    *(uint4*)&Bs[buf][r][c0 + 8] = b1;
  };

  // Wave mapping: 8 waves arranged 4(M) x 2(N); wave tile 32x64 = 2x4 WMMA tiles.
  const int lane = tid & 31;
  const int half = lane >> 4;    // K-half selector per ISA 16-bit A/B layouts
  const int lrow = lane & 15;
  const int wave = tid >> 5;
  const int wm   = wave >> 1;    // 0..3
  const int wn   = wave & 1;     // 0..1

  v8f acc[2][4] = {};

  constexpr int KITER = IN_F / KTILE;   // 128
  load_tile(0);
  for (int kt = 0; kt < KITER; ++kt) {
    const int buf = kt & 1;
    store_tile(buf);
    __syncthreads();

    if (kt + 1 < KITER) {
      load_tile(kt + 1);                 // overlap global loads with WMMA phase
      if (kt + 2 < KITER) {              // global_prefetch_b8 two tiles ahead
        __builtin_prefetch(x  + x_row + (size_t)(kt + 2) * KTILE + c0, 0, 0);
        __builtin_prefetch(Wq + q_row + (size_t)(kt + 2) * KTILE + c0, 0, 0);
      }
    }

    // A fragments: lane row = lane&15, K = half*8 + {0..7} and +16 (ISA 16-bit A layout)
    Frag a[2], b[4];
    #pragma unroll
    for (int tm = 0; tm < 2; ++tm) {
      const unsigned short* p = &As[buf][wm * 32 + tm * 16 + lrow][half * 8];
      const uint4 lo = *(const uint4*)p;
      const uint4 hi = *(const uint4*)(p + 16);
      a[tm].u[0] = lo.x; a[tm].u[1] = lo.y; a[tm].u[2] = lo.z; a[tm].u[3] = lo.w;
      a[tm].u[4] = hi.x; a[tm].u[5] = hi.y; a[tm].u[6] = hi.z; a[tm].u[7] = hi.w;
    }
    // B fragments: lane col = lane&15, K = half*8 + {0..7} and +16
    #pragma unroll
    for (int tn = 0; tn < 4; ++tn) {
      const unsigned short* p = &Bs[buf][wn * 64 + tn * 16 + lrow][half * 8];
      const uint4 lo = *(const uint4*)p;
      const uint4 hi = *(const uint4*)(p + 16);
      b[tn].u[0] = lo.x; b[tn].u[1] = lo.y; b[tn].u[2] = lo.z; b[tn].u[3] = lo.w;
      b[tn].u[4] = hi.x; b[tn].u[5] = hi.y; b[tn].u[6] = hi.z; b[tn].u[7] = hi.w;
    }

    #pragma unroll
    for (int tm = 0; tm < 2; ++tm)
      #pragma unroll
      for (int tn = 0; tn < 4; ++tn)
        acc[tm][tn] = __builtin_amdgcn_wmma_f32_16x16x32_bf16(
            false, a[tm].v, false, b[tn].v, (short)0, acc[tm][tn], false, false);

    __syncthreads();
  }

  // Epilogue: C/D layout — row = 8*(lane>>4) + v, col = lane&15. Add bias, store fp32.
  #pragma unroll
  for (int tn = 0; tn < 4; ++tn) {
    const int n  = n_base + wn * 64 + tn * 16 + lrow;
    const float bv = bias[n];
    #pragma unroll
    for (int tm = 0; tm < 2; ++tm) {
      const int m0 = m_base + wm * 32 + tm * 16 + half * 8;
      #pragma unroll
      for (int v = 0; v < 8; ++v) {
        out[(size_t)(m0 + v) * OUT_F + n] = acc[tm][tn][v] + bv;
      }
    }
  }
}

extern "C" void kernel_launch(void* const* d_in, const int* in_sizes, int n_in,
                              void* d_out, int out_size, void* d_ws, size_t ws_size,
                              hipStream_t stream) {
  (void)in_sizes; (void)n_in; (void)out_size; (void)d_ws; (void)ws_size;
  const float* x     = (const float*)d_in[0];
  const int*   Wq    = (const int*)  d_in[1];
  const float* scale = (const float*)d_in[2];
  const float* zero  = (const float*)d_in[3];
  const float* bias  = (const float*)d_in[4];
  float*       out   = (float*)d_out;

  dim3 grid(OUT_F / NT, MTOT / MT);   // (32, 128); x-major dispatch keeps x strips hot in L2
  dim3 block(256);
  qlinear_wmma_bf16<<<grid, block, 0, stream>>>(x, Wq, scale, zero, bias, out);
}
